// RNN_77137612636244
// MI455X (gfx1250) — compile-verified, matrix operand order checked by reading, not run on hardware
//
#include <hip/hip_runtime.h>
#include <hip/hip_bf16.h>
#include <cstdint>

// ---------------------------------------------------------------------------
// RNN LM forward for MI455X (gfx1250, wave32, WMMA + async-to-LDS pipeline).
// B=16 S=256 V=32000 I=H=1024 L=2
// ---------------------------------------------------------------------------

typedef __bf16 bf16;
typedef bf16  v16bf __attribute__((ext_vector_type(16)));
typedef bf16  v8bf  __attribute__((ext_vector_type(8)));
typedef float v8f   __attribute__((ext_vector_type(8)));

#define RB 16      // batch
#define RS 256     // seq len
#define RV 32000   // vocab
#define RI 1024    // input dim
#define RH 1024    // hidden dim
#define RL 2       // layers
#define LNEPS 1e-5f

#define ROWE 40    // padded LDS row stride in bf16 elems (80B; avoids bank conflicts)

// --- CDNA5 async copy: LDS[vdst] = MEM[vaddr], 16B per lane (ASYNCcnt) -----
__device__ __forceinline__ void async_copy_b128(uint32_t lds_addr, const bf16* g) {
  asm volatile("global_load_async_to_lds_b128 %0, %1, off"
               :: "v"(lds_addr), "v"((uint64_t)(uintptr_t)g)
               : "memory");
}

#if __has_builtin(__builtin_amdgcn_s_wait_asynccnt)
#define WAIT_ASYNC(n) __builtin_amdgcn_s_wait_asynccnt(n)
#else
#define WAIT_ASYNC(n) asm volatile("s_wait_asynccnt %0" :: "i"(n) : "memory")
#endif

// --- WMMA bf16 fragment loaders (ISA 7.12.2 layout) ------------------------
// lane-half h: elements 0..7  <- K = k0 + 8h .. k0+8h+7
//              elements 8..15 <- K = k0 + 16 + 8h ..
__device__ __forceinline__ v16bf load_frag_row(const bf16* __restrict__ row,
                                               int k0, int half) {
  const v8bf lo = *(const v8bf*)(row + k0 + half * 8);
  const v8bf hi = *(const v8bf*)(row + k0 + 16 + half * 8);
  v16bf f;
#pragma unroll
  for (int i = 0; i < 8; ++i) { f[i] = lo[i]; f[i + 8] = hi[i]; }
  return f;
}

// fragment from an LDS k-panel with padded row stride ROWE
__device__ __forceinline__ v16bf lds_frag(const bf16* __restrict__ buf,
                                          int tile16, int l16, int half) {
  const bf16* row = buf + (tile16 * 16 + l16) * ROWE;
  const v8bf lo = *(const v8bf*)(row + half * 8);
  const v8bf hi = *(const v8bf*)(row + 16 + half * 8);
  v16bf f;
#pragma unroll
  for (int i = 0; i < 8; ++i) { f[i] = lo[i]; f[i + 8] = hi[i]; }
  return f;
}

// ---------------------------------------------------------------------------
// C[M,N] = A[M,K](bf16) * B[N,K](bf16)^T + bias[N], fp32 out.
// Block: 256 threads = 8 waves. Block tile 64(M) x 256(N), k-chunk 32.
// Wave: 2 M-tiles x 4 N-tiles (8 v8f accumulators).
// A/B k-panels double-buffered in LDS via global_load_async_to_lds_b128:
//   per stage 320 rows x 64B = 40 b128-copies = 5 per wave.
// ---------------------------------------------------------------------------
__device__ __forceinline__ void stage_issue(const bf16* __restrict__ Ag,
                                            const bf16* __restrict__ Bg,
                                            int K, int k0,
                                            uint32_t aLds, uint32_t bLds,
                                            int wave, int lane) {
  const int rsub = lane >> 2;      // 8 rows per copy instruction
  const int csub = lane & 3;       // 4 x 16B per 64B row
#pragma unroll
  for (int j = 0; j < 5; ++j) {
    const int q = wave * 5 + j;    // 0..39 (wave-uniform)
    if (q < 8) {                   // A panel: 64 rows
      const int r = q * 8 + rsub;
      async_copy_b128(aLds + (uint32_t)(r * (ROWE * 2) + csub * 16),
                      Ag + (size_t)r * K + k0 + csub * 8);
    } else {                       // B panel: 256 rows
      const int r = (q - 8) * 8 + rsub;
      async_copy_b128(bLds + (uint32_t)(r * (ROWE * 2) + csub * 16),
                      Bg + (size_t)r * K + k0 + csub * 8);
    }
  }
}

__global__ __launch_bounds__(256) void gemm_bf16_tn(
    const bf16* __restrict__ A, const bf16* __restrict__ Bm,
    const float* __restrict__ bias, float* __restrict__ C,
    int M, int N, int K) {
  __shared__ bf16 As[2][64 * ROWE];    // 2 x 5.0 KB
  __shared__ bf16 Bs[2][256 * ROWE];   // 2 x 20.0 KB

  const int lane = threadIdx.x & 31;
  const int wave = threadIdx.x >> 5;
  const int half = lane >> 4;
  const int l16  = lane & 15;
  const int wm   = wave & 1;           // m-half of block tile
  const int wn   = wave >> 1;          // n-quarter of block tile

  const int m_block = blockIdx.x * 64;
  const int n_block = blockIdx.y * 256;

  const bf16* Ag = A + (size_t)m_block * K;
  const bf16* Bg = Bm + (size_t)n_block * K;

  const uint32_t aLds[2] = { (uint32_t)(uintptr_t)&As[0][0],
                             (uint32_t)(uintptr_t)&As[1][0] };
  const uint32_t bLds[2] = { (uint32_t)(uintptr_t)&Bs[0][0],
                             (uint32_t)(uintptr_t)&Bs[1][0] };

  v8f acc[2][4];
#pragma unroll
  for (int mi = 0; mi < 2; ++mi)
#pragma unroll
    for (int ni = 0; ni < 4; ++ni) acc[mi][ni] = (v8f){};

  const int KSTEPS = K >> 5;
  stage_issue(Ag, Bg, K, 0, aLds[0], bLds[0], wave, lane);

  for (int ks = 0; ks < KSTEPS; ++ks) {
    const int cur = ks & 1;
    if (ks + 1 < KSTEPS) {
      stage_issue(Ag, Bg, K, (ks + 1) << 5, aLds[cur ^ 1], bLds[cur ^ 1],
                  wave, lane);
      WAIT_ASYNC(5);                   // own previous-stage copies complete
    } else {
      WAIT_ASYNC(0);
    }
    __syncthreads();                   // all waves' copies visible in LDS

    const bf16* Ab = As[cur];
    const bf16* Bb = Bs[cur];
    const v16bf fa0 = lds_frag(Ab, wm * 2 + 0, l16, half);
    const v16bf fa1 = lds_frag(Ab, wm * 2 + 1, l16, half);
    const v16bf fb0 = lds_frag(Bb, wn * 4 + 0, l16, half);
    const v16bf fb1 = lds_frag(Bb, wn * 4 + 1, l16, half);
    const v16bf fb2 = lds_frag(Bb, wn * 4 + 2, l16, half);
    const v16bf fb3 = lds_frag(Bb, wn * 4 + 3, l16, half);
    acc[0][0] = __builtin_amdgcn_wmma_f32_16x16x32_bf16(false, fa0, false, fb0, (short)0, acc[0][0], false, false);
    acc[0][1] = __builtin_amdgcn_wmma_f32_16x16x32_bf16(false, fa0, false, fb1, (short)0, acc[0][1], false, false);
    acc[0][2] = __builtin_amdgcn_wmma_f32_16x16x32_bf16(false, fa0, false, fb2, (short)0, acc[0][2], false, false);
    acc[0][3] = __builtin_amdgcn_wmma_f32_16x16x32_bf16(false, fa0, false, fb3, (short)0, acc[0][3], false, false);
    acc[1][0] = __builtin_amdgcn_wmma_f32_16x16x32_bf16(false, fa1, false, fb0, (short)0, acc[1][0], false, false);
    acc[1][1] = __builtin_amdgcn_wmma_f32_16x16x32_bf16(false, fa1, false, fb1, (short)0, acc[1][1], false, false);
    acc[1][2] = __builtin_amdgcn_wmma_f32_16x16x32_bf16(false, fa1, false, fb2, (short)0, acc[1][2], false, false);
    acc[1][3] = __builtin_amdgcn_wmma_f32_16x16x32_bf16(false, fa1, false, fb3, (short)0, acc[1][3], false, false);
    __syncthreads();                   // protect buffer for next-next stage
  }

#pragma unroll
  for (int mi = 0; mi < 2; ++mi) {
    const int mt = m_block + (wm * 2 + mi) * 16;
#pragma unroll
    for (int v = 0; v < 8; ++v) {
      float* crow = C + (size_t)(mt + v + 8 * half) * N;  // D: M = v + 8*half
#pragma unroll
      for (int ni = 0; ni < 4; ++ni) {
        const int n = n_block + (wn * 4 + ni) * 16 + l16;
        crow[n] = acc[mi][ni][v] + bias[n];
      }
    }
  }
}

// --- helpers ---------------------------------------------------------------
__global__ void f32_to_bf16_k(const float* __restrict__ src,
                              bf16* __restrict__ dst, size_t n) {
  size_t i = (size_t)blockIdx.x * blockDim.x + threadIdx.x;
  if (i < n) dst[i] = (bf16)src[i];
}

__global__ void embed_gather_k(const int* __restrict__ tokens,
                               const float* __restrict__ emb,
                               bf16* __restrict__ xbf) {
  const int row = blockIdx.x;              // row = b*S + s
  const int tok = tokens[row];
  const float* src = emb + (size_t)tok * RI;
  bf16* dst = xbf + (size_t)row * RI;
  for (int i = threadIdx.x; i < RI; i += blockDim.x) dst[i] = (bf16)src[i];
}

__global__ void zero_state_k(bf16* __restrict__ h0, bf16* __restrict__ h1) {
  int i = blockIdx.x * blockDim.x + threadIdx.x;
  if (i < RB * RH) { h0[i] = (bf16)0.0f; h1[i] = (bf16)0.0f; }
}

// --- LayerNorm + tanh over 16 rows of H=1024 (1024 threads) ---------------
__device__ __forceinline__ void ln_tanh_rows(
    const float* __restrict__ pre, const float* __restrict__ g,
    const float* __restrict__ b, bf16* __restrict__ hs,
    float* __restrict__ hT_l,            // fp32 out for last step, else null
    float* psum, float* psq, float* rmu, float* rinv) {
  const int tid = threadIdx.x;
  const int row = tid >> 6;              // 16 rows, 64 threads/row
  const int sub = tid & 63;              // each thread: 16 contiguous elems
  const float* pr = pre + row * RH + sub * 16;
  float s = 0.f, q = 0.f;
#pragma unroll
  for (int i = 0; i < 16; ++i) { float x = pr[i]; s += x; q += x * x; }
  psum[tid] = s; psq[tid] = q;
  __syncthreads();
  if (tid < 16) {
    float ts = 0.f, tq = 0.f;
    for (int j = 0; j < 64; ++j) { ts += psum[tid * 64 + j]; tq += psq[tid * 64 + j]; }
    const float mu  = ts * (1.0f / RH);
    const float var = tq * (1.0f / RH) - mu * mu;
    rmu[tid]  = mu;
    rinv[tid] = rsqrtf(var + LNEPS);
  }
  __syncthreads();
  const float mu = rmu[row], inv = rinv[row];
  bf16* hd = hs + row * RH + sub * 16;
#pragma unroll
  for (int i = 0; i < 16; ++i) {
    const int n = sub * 16 + i;
    const float y = tanhf((pr[i] - mu) * inv * g[n] + b[n]);
    hd[i] = (bf16)y;
    if (hT_l) hT_l[row * RH + n] = y;
  }
  __syncthreads();
}

// ---------------------------------------------------------------------------
// One timestep, both layers. Single workgroup, 1024 threads = 32 waves.
// pre[b, :] = h0 @ w_hh0^T + xw0[b*S+t, :] + b_hh0   -> h0 = tanh(LN(pre))
// pre[b, :] = h0 @ w_ih1^T + h1 @ w_hh1^T + biases   -> h1 = tanh(LN(pre))
// h state kept in LDS (bf16) during the step; persisted to ws between steps.
// Weights (bf16, 2MB each) stream from L2; each row read by exactly one wave.
// ---------------------------------------------------------------------------
__global__ __launch_bounds__(1024) void rnn_step_k(
    const float* __restrict__ xw0,   // [B*S, H] incl. b_ih0
    const bf16* __restrict__ w_hh0,  // [H,H]
    const bf16* __restrict__ w_ih1,  // [H,H]
    const bf16* __restrict__ w_hh1,  // [H,H]
    const float* __restrict__ b_ih,  // [L,H]
    const float* __restrict__ b_hh,  // [L,H]
    const float* __restrict__ ln_g,  // [L,H]
    const float* __restrict__ ln_b,  // [L,H]
    bf16* __restrict__ h0_st, bf16* __restrict__ h1_st,  // [B*H]
    bf16* __restrict__ outs,         // [B*S, H]
    float* __restrict__ hT,          // [L,B,H] (d_out tail) or null
    int t) {
  extern __shared__ char smem[];
  bf16*  h0s  = (bf16*)smem;                         // 32 KB
  bf16*  h1s  = (bf16*)(smem + 32768);               // 32 KB
  float* pre  = (float*)(smem + 65536);              // 64 KB
  float* psum = (float*)(smem + 131072);             // 4 KB
  float* psq  = psum + 1024;                         // 4 KB
  float* rmu  = psq + 1024;                          // 64 B
  float* rinv = rmu + 16;                            // 64 B

  const int tid  = threadIdx.x;
  const int lane = tid & 31;
  const int wave = tid >> 5;
  const int half = lane >> 4;
  const int l16  = lane & 15;

  for (int i = tid; i < RB * RH; i += 1024) { h0s[i] = h0_st[i]; h1s[i] = h1_st[i]; }
  __syncthreads();

  // ---- layer 0: hidden GEMM (M=16, N=1024, K=1024) ----
  for (int it = 0; it < 2; ++it) {
    const int nb = (wave + it * 32) * 16;
    const bf16* arow = h0s + (size_t)l16 * RH;
    const bf16* brow = w_hh0 + (size_t)(nb + l16) * RH;
    v8f acc = {};
    for (int k0 = 0; k0 < RH; k0 += 32) {
      v16bf a = load_frag_row(arow, k0, half);
      v16bf b = load_frag_row(brow, k0, half);
      acc = __builtin_amdgcn_wmma_f32_16x16x32_bf16(false, a, false, b, (short)0, acc, false, false);
    }
#pragma unroll
    for (int v = 0; v < 8; ++v) {
      const int m = v + 8 * half;
      const int n = nb + l16;
      pre[m * RH + n] = acc[v] + xw0[((size_t)m * RS + t) * RH + n] + b_hh[n];
    }
  }
  __syncthreads();

  ln_tanh_rows(pre, ln_g, ln_b, h0s, hT ? hT : nullptr, psum, psq, rmu, rinv);

  // ---- layer 1: dual GEMM (input from new h0, recurrent from h1) ----
  for (int it = 0; it < 2; ++it) {
    const int nb = (wave + it * 32) * 16;
    const bf16* a0row = h0s + (size_t)l16 * RH;
    const bf16* a1row = h1s + (size_t)l16 * RH;
    const bf16* b0row = w_ih1 + (size_t)(nb + l16) * RH;
    const bf16* b1row = w_hh1 + (size_t)(nb + l16) * RH;
    v8f acc = {};
    for (int k0 = 0; k0 < RH; k0 += 32) {
      v16bf a0 = load_frag_row(a0row, k0, half);
      v16bf b0 = load_frag_row(b0row, k0, half);
      acc = __builtin_amdgcn_wmma_f32_16x16x32_bf16(false, a0, false, b0, (short)0, acc, false, false);
      v16bf a1 = load_frag_row(a1row, k0, half);
      v16bf b1 = load_frag_row(b1row, k0, half);
      acc = __builtin_amdgcn_wmma_f32_16x16x32_bf16(false, a1, false, b1, (short)0, acc, false, false);
    }
#pragma unroll
    for (int v = 0; v < 8; ++v) {
      const int m = v + 8 * half;
      const int n = nb + l16;
      pre[m * RH + n] = acc[v] + b_ih[RH + n] + b_hh[RH + n];
    }
  }
  __syncthreads();

  ln_tanh_rows(pre, ln_g + RH, ln_b + RH, h1s, hT ? hT + RB * RH : nullptr,
               psum, psq, rmu, rinv);

  // ---- persist state + emit outs[:, t, :] ----
  for (int i = tid; i < RB * RH; i += 1024) {
    const int m = i >> 10, n = i & (RH - 1);
    outs[((size_t)m * RS + t) * RH + n] = h1s[i];
    h0_st[i] = h0s[i];
    h1_st[i] = h1s[i];
  }
}

// ---------------------------------------------------------------------------
extern "C" void kernel_launch(void* const* d_in, const int* in_sizes, int n_in,
                              void* d_out, int out_size, void* d_ws, size_t ws_size,
                              hipStream_t stream) {
  const int*   tokens = (const int*)d_in[0];
  const float* emb    = (const float*)d_in[1];
  const float* w_ih   = (const float*)d_in[2];
  const float* w_hh   = (const float*)d_in[3];
  const float* b_ih   = (const float*)d_in[4];
  const float* b_hh   = (const float*)d_in[5];
  const float* ln_g   = (const float*)d_in[6];
  const float* ln_b   = (const float*)d_in[7];
  const float* h2o_w  = (const float*)d_in[8];
  const float* h2o_b  = (const float*)d_in[9];
  // d_in[10] = h0 (all zeros) -- states are zero-initialized on device.

  // ---- workspace layout (~103 MB) ----
  char* ws = (char*)d_ws;
  size_t off = 0;
  auto carve = [&](size_t bytes) {
    void* p = ws + off;
    off += (bytes + 255) & ~(size_t)255;
    return p;
  };
  bf16*  x_bf    = (bf16*) carve((size_t)RB * RS * RI * 2);
  float* xw0     = (float*)carve((size_t)RB * RS * RH * 4);
  bf16*  wih_bf  = (bf16*) carve((size_t)RL * RH * RI * 2);
  bf16*  whh_bf  = (bf16*) carve((size_t)RL * RH * RH * 2);
  bf16*  h2o_bf  = (bf16*) carve((size_t)RV * RH * 2);
  bf16*  outs_bf = (bf16*) carve((size_t)RB * RS * RH * 2);
  bf16*  h0_st   = (bf16*) carve((size_t)RB * RH * 2);
  bf16*  h1_st   = (bf16*) carve((size_t)RB * RH * 2);

  float* logits = (float*)d_out;                      // [B,S,V]
  float* hT     = logits + (size_t)RB * RS * RV;      // [L,B,H]

  // 1) weights -> bf16
  {
    size_t n = (size_t)RL * RH * RI;
    f32_to_bf16_k<<<(unsigned)((n + 255) / 256), 256, 0, stream>>>(w_ih, wih_bf, n);
    n = (size_t)RL * RH * RH;
    f32_to_bf16_k<<<(unsigned)((n + 255) / 256), 256, 0, stream>>>(w_hh, whh_bf, n);
    n = (size_t)RV * RH;
    f32_to_bf16_k<<<(unsigned)((n + 255) / 256), 256, 0, stream>>>(h2o_w, h2o_bf, n);
  }

  // 2) embedding gather -> bf16 x [B*S, I]
  embed_gather_k<<<RB * RS, 256, 0, stream>>>(tokens, emb, x_bf);

  // 3) xw0 = x @ w_ih0^T + b_ih0  (time-parallel input projection, WMMA)
  {
    dim3 grid(RB * RS / 64, RH / 256);
    gemm_bf16_tn<<<grid, 256, 0, stream>>>(x_bf, wih_bf, b_ih, xw0,
                                           RB * RS, RH, RI);
  }

  // 4) recurrent scan (sequential over S)
  zero_state_k<<<(RB * RH + 255) / 256, 256, 0, stream>>>(h0_st, h1_st);
  const size_t smem_bytes = 32768 + 32768 + 65536 + 4096 + 4096 + 128;
  for (int t = 0; t < RS; ++t) {
    rnn_step_k<<<1, 1024, smem_bytes, stream>>>(
        xw0, whh_bf, wih_bf + (size_t)RH * RI, whh_bf + (size_t)RH * RH,
        b_ih, b_hh, ln_g, ln_b, h0_st, h1_st, outs_bf,
        (t == RS - 1) ? hT : nullptr, t);
  }

  // 5) logits = outs @ h2o_w^T + h2o_b  (the big WMMA GEMM: 268 GFLOP)
  {
    dim3 grid(RB * RS / 64, RV / 256);
    gemm_bf16_tn<<<grid, 256, 0, stream>>>(outs_bf, h2o_bf, h2o_b, logits,
                                           RB * RS, RV, RH);
  }
}
